// PrototypePrediction_64750926954869
// MI455X (gfx1250) — matrix operand order repaired and checked by reading, not compiled, not run on hardware
//
#include <hip/hip_runtime.h>
#include <hip/hip_bf16.h>

// ---------------------------------------------------------------------------
// PrototypePrediction fused kernel for gfx1250 (MI455X, wave32, WMMA).
//
// Reference:  xn = l2norm(x); sim = xn @ pn^T; prob = softmax(sim);
//             out = (prob @ P)/K @ W^T + b
// Fusions:    PW  = P @ W^T / K   (precomputed, tiny)  -> out = prob @ PW + b
//             sim = (x @ pn^T) * (1/||x||)             -> single pass over x
// ---------------------------------------------------------------------------

typedef __attribute__((ext_vector_type(16))) _Float16 v16h;
typedef __attribute__((ext_vector_type(8)))  _Float16 v8h;
typedef __attribute__((ext_vector_type(4)))  _Float16 v4h;
typedef __attribute__((ext_vector_type(8)))  float    v8f;

#define D_DIM   768
#define K_PROTO 256
#define ROWS    64      // rows of x per block
#define CHUNK   128     // D-chunk staged in LDS per step

union AFrag { v16h v; v8h h[2]; };

static __device__ __forceinline__ v8f wmma16(v16h a, v16h b, v8f c) {
  // v_wmma_f32_16x16x32_f16 : D = A(16x32) * B(32x16) + C(16x16 f32)
  return __builtin_amdgcn_wmma_f32_16x16x32_f16(false, a, false, b,
                                                (short)0, c, false, false);
}

// ---------------------------------------------------------------------------
// Prep 1: per-prototype L2 norm -> pn (f16, row-major), raw protos -> ph (f16)
// ---------------------------------------------------------------------------
__global__ __launch_bounds__(256)
void prep_protos(const float* __restrict__ protos,
                 _Float16* __restrict__ pn,
                 _Float16* __restrict__ ph) {
  __shared__ float ws[8];
  __shared__ float s_inv;
  int k = blockIdx.x, tid = threadIdx.x;
  const float* pr = protos + k * D_DIM;
  float s = 0.f;
  for (int i = tid; i < D_DIM; i += 256) { float v = pr[i]; s += v * v; }
#pragma unroll
  for (int m = 16; m >= 1; m >>= 1) s += __shfl_xor(s, m, 32);
  if ((tid & 31) == 0) ws[tid >> 5] = s;
  __syncthreads();
  if (tid == 0) {
    float t = 0.f;
#pragma unroll
    for (int i = 0; i < 8; ++i) t += ws[i];
    s_inv = 1.0f / fmaxf(sqrtf(t), 1e-12f);
  }
  __syncthreads();
  float inv = s_inv;
  for (int i = tid; i < D_DIM; i += 256) {
    float v = pr[i];
    pn[k * D_DIM + i] = (_Float16)(v * inv);
    ph[k * D_DIM + i] = (_Float16)v;
  }
}

// ---------------------------------------------------------------------------
// Prep 2: W (f32) -> wh (f16), same row-major layout (W[n][d]).
// ---------------------------------------------------------------------------
__global__ __launch_bounds__(256)
void prep_w(const float* __restrict__ W, _Float16* __restrict__ wh, int n) {
  int i = blockIdx.x * 256 + threadIdx.x;
  if (i < n) wh[i] = (_Float16)W[i];
}

// ---------------------------------------------------------------------------
// Prep 3: PW[k][n] = (1/K) * sum_d P[k][d] * W[n][d], stored transposed:
//         pwt[n][k]  (so GEMM-B reads are k-contiguous per lane).
// One 16x16 tile per wave, WMMA f16->f32. Grid = 16*48/4 blocks of 128 thr.
// ---------------------------------------------------------------------------
__global__ __launch_bounds__(128)
void prep_pw(const _Float16* __restrict__ ph,
             const _Float16* __restrict__ wh,
             _Float16* __restrict__ pwt) {
  int lane = threadIdx.x & 31, wave = threadIdx.x >> 5;
  int lm = lane & 15, lh = lane >> 4;
  int tile = blockIdx.x * 4 + wave;
  int kt = (tile & 15) * 16;   // M base: prototype index
  int nt = (tile >> 4) * 16;   // N base: output-feature index
  v8f acc = {};
  for (int d0 = 0; d0 < D_DIM; d0 += 32) {
    AFrag A;
    const _Float16* ab = ph + (kt + lm) * D_DIM + d0 + lh * 16;
    A.h[0] = *(const v8h*)ab;
    A.h[1] = *(const v8h*)(ab + 8);
    v16h B = *(const v16h*)(wh + (nt + lm) * D_DIM + d0 + lh * 16);
    acc = wmma16(A.v, B, acc);
  }
  const float invK = 1.0f / (float)K_PROTO;
#pragma unroll
  for (int r = 0; r < 8; ++r) {
    int kk = kt + r + lh * 8;
    int nn = nt + lm;
    pwt[nn * K_PROTO + kk] = (_Float16)(acc[r] * invK);
  }
}

// ---------------------------------------------------------------------------
// Main fused kernel: 128 threads = 4 waves; 64 rows of x per block
// (16 rows per wave). Two WMMA GEMMs with in-register softmax between.
// Single pass over x: sum-of-squares accumulated during f16 staging,
// normalization applied as a per-row scalar on sim before softmax.
// ---------------------------------------------------------------------------
__global__ __launch_bounds__(128)
void proto_main(const float*    __restrict__ x,     // (BN, 768) f32
                const _Float16* __restrict__ pn,    // (256, 768) normalized
                const _Float16* __restrict__ pwt,   // (768, 256) = (P@W^T/K)^T
                const float*    __restrict__ bias,  // (768)
                float*          __restrict__ out) { // (BN, 768) f32
  __shared__ float s_sq[ROWS];
  __shared__ __align__(32) _Float16 s_buf[4 * 16 * K_PROTO]; // 32 KB (union use)

  const int tid  = threadIdx.x;
  const int lane = tid & 31, wave = tid >> 5;
  const int lm = lane & 15, lh = lane >> 4;
  const int rowbase = blockIdx.x * ROWS;

  if (tid < ROWS) s_sq[tid] = 0.f;

  // ---- Phase 1: sim = x @ pn^T  (16 rows x 256 cols per wave) ------------
  v8f zero = {};
  v8f acc[16];
#pragma unroll
  for (int t = 0; t < 16; ++t) acc[t] = zero;

  for (int c0 = 0; c0 < D_DIM; c0 += CHUNK) {
    __syncthreads();
    // stage raw x chunk as f16 (ROWS x CHUNK); accumulate row sum-of-squares
    const bool pf = (c0 + CHUNK < D_DIM);
    for (int i = tid * 4; i < ROWS * CHUNK; i += 128 * 4) {
      int r = i >> 7, col = i & (CHUNK - 1);
      const float* xp = x + (rowbase + r) * D_DIM + c0 + col;
      float4 v = *(const float4*)xp;
      atomicAdd(&s_sq[r], v.x * v.x + v.y * v.y + v.z * v.z + v.w * v.w);
      v4h h;
      h[0] = (_Float16)v.x;
      h[1] = (_Float16)v.y;
      h[2] = (_Float16)v.z;
      h[3] = (_Float16)v.w;
      *(v4h*)&s_buf[i] = h;
      // one prefetch per 128B line of the next chunk (col is uniform/thread)
      if (pf && (col & 31) == 0) __builtin_prefetch(xp + CHUNK, 0, 1);
    }
    __syncthreads();
#pragma unroll
    for (int kk = 0; kk < CHUNK; kk += 32) {
      AFrag A;
      const _Float16* ab = &s_buf[(wave * 16 + lm) * CHUNK + kk + lh * 16];
      A.h[0] = *(const v8h*)ab;
      A.h[1] = *(const v8h*)(ab + 8);
#pragma unroll
      for (int t = 0; t < 16; ++t) {
        v16h B = *(const v16h*)(pn + (t * 16 + lm) * D_DIM + c0 + kk + lh * 16);
        acc[t] = wmma16(A.v, B, acc[t]);
      }
    }
  }

  // ---- Phase 2: scale rows by 1/||x||, softmax over 256 cols -------------
  // C layout: lane holds rows r + 8*lh (r=0..7); each row's 256 values live
  // in 16 tiles x 16 lanes of the same half -> shfl_xor masks 1,2,4,8.
  // (sum-of-squares atomics completed before the last compute barrier.)
  float invsum[8];
#pragma unroll
  for (int r = 0; r < 8; ++r) {
    float inv = 1.0f / fmaxf(sqrtf(s_sq[wave * 16 + r + lh * 8]), 1e-12f);
    float m = acc[0][r] * inv;
#pragma unroll
    for (int t = 0; t < 16; ++t) {
      float v = acc[t][r] * inv;
      acc[t][r] = v;
      m = fmaxf(m, v);
    }
    m = fmaxf(m, __shfl_xor(m, 1, 32));
    m = fmaxf(m, __shfl_xor(m, 2, 32));
    m = fmaxf(m, __shfl_xor(m, 4, 32));
    m = fmaxf(m, __shfl_xor(m, 8, 32));
    float s = 0.f;
#pragma unroll
    for (int t = 0; t < 16; ++t) {
      float e = __expf(acc[t][r] - m);
      acc[t][r] = e;
      s += e;
    }
    s += __shfl_xor(s, 1, 32);
    s += __shfl_xor(s, 2, 32);
    s += __shfl_xor(s, 4, 32);
    s += __shfl_xor(s, 8, 32);
    invsum[r] = 1.0f / s;
  }

  // ---- Phase 3: prob (f16) -> LDS, row-major per wave --------------------
  __syncthreads();  // all waves done reading s_buf before reuse
  _Float16* pb = &s_buf[wave * 16 * K_PROTO];
#pragma unroll
  for (int r = 0; r < 8; ++r) {
    int row = r + lh * 8;
#pragma unroll
    for (int t = 0; t < 16; ++t)
      pb[row * K_PROTO + t * 16 + lm] = (_Float16)(acc[t][r] * invsum[r]);
  }
  __syncthreads();

  // ---- Phase 4: out = prob @ pwt(view) + b  (1/K folded into pwt) --------
  for (int n0 = 0; n0 < D_DIM; n0 += 128) {
    v8f acc2[8];
#pragma unroll
    for (int t = 0; t < 8; ++t) acc2[t] = zero;
#pragma unroll
    for (int kk = 0; kk < K_PROTO; kk += 32) {
      AFrag A;
      const _Float16* ab = pb + lm * K_PROTO + kk + lh * 16;
      A.h[0] = *(const v8h*)ab;
      A.h[1] = *(const v8h*)(ab + 8);
#pragma unroll
      for (int t = 0; t < 8; ++t) {
        v16h B = *(const v16h*)(pwt + (n0 + t * 16 + lm) * K_PROTO + kk + lh * 16);
        acc2[t] = wmma16(A.v, B, acc2[t]);
      }
    }
#pragma unroll
    for (int t = 0; t < 8; ++t) {
      int col = n0 + t * 16 + lm;
      float bv = bias[col];
#pragma unroll
      for (int r = 0; r < 8; ++r) {
        int row = rowbase + wave * 16 + r + lh * 8;
        out[row * D_DIM + col] = acc2[t][r] + bv;
      }
    }
  }
}

// ---------------------------------------------------------------------------
extern "C" void kernel_launch(void* const* d_in, const int* in_sizes, int n_in,
                              void* d_out, int out_size, void* d_ws, size_t ws_size,
                              hipStream_t stream) {
  const float* x      = (const float*)d_in[0];  // (B,N,768)
  const float* protos = (const float*)d_in[1];  // (256,768)
  const float* W      = (const float*)d_in[2];  // (768,768)
  const float* b      = (const float*)d_in[3];  // (768)
  float* out = (float*)d_out;

  char* ws = (char*)d_ws;
  _Float16* pn  = (_Float16*)(ws + 0);        // 256*768*2 = 393216
  _Float16* ph  = (_Float16*)(ws + 393216);   // 256*768*2
  _Float16* wh  = (_Float16*)(ws + 786432);   // 768*768*2 = 1179648
  _Float16* pwt = (_Float16*)(ws + 1966080);  // 768*256*2 = 393216

  int BN = in_sizes[0] / D_DIM;               // 131072

  prep_protos<<<K_PROTO, 256, 0, stream>>>(protos, pn, ph);
  prep_w<<<(D_DIM * D_DIM + 255) / 256, 256, 0, stream>>>(W, wh, D_DIM * D_DIM);
  prep_pw<<<(K_PROTO / 16) * (D_DIM / 16) / 4, 128, 0, stream>>>(ph, wh, pwt);
  proto_main<<<BN / ROWS, 128, 0, stream>>>(x, pn, pwt, b, out);
}
